// RIDecoder_68487548502108
// MI455X (gfx1250) — matrix operand order repaired and checked by reading, not compiled, not run on hardware
//
#include <hip/hip_runtime.h>
#include <hip/hip_bf16.h>
#include <math.h>

typedef unsigned short ushort_t;
typedef __attribute__((ext_vector_type(16))) __bf16 bf16x16;
typedef __attribute__((ext_vector_type(8)))  float  f32x8;
typedef unsigned int u32x4 __attribute__((ext_vector_type(4)));

union FragU { bf16x16 v; ushort_t u[16]; u32x4 q[2]; };
static_assert(sizeof(FragU) == 32, "frag size");

// ---------------- bf16 helpers ----------------
__device__ __forceinline__ ushort_t f2bf(float f) {
  union { float f; unsigned int u; } v; v.f = f;
  unsigned int u = v.u;
  unsigned int r = (u + 0x7FFFu + ((u >> 16) & 1u)) >> 16;   // round-nearest-even
  return (ushort_t)r;
}
__device__ __forceinline__ float bf2f(ushort_t h) {
  union { unsigned int u; float f; } v; v.u = ((unsigned int)h) << 16;
  return v.f;
}
__device__ __forceinline__ f32x8 zero8() {
  f32x8 z;
#pragma unroll
  for (int i = 0; i < 8; ++i) z[i] = 0.f;
  return z;
}
__device__ __forceinline__ f32x8 wmma_bf16(bf16x16 a, bf16x16 b, f32x8 c) {
  return __builtin_amdgcn_wmma_f32_16x16x32_bf16(false, a, false, b, (short)0, c, false, false);
}

// ---- A fragment from global row-major bf16 [.. x lda], two contiguous 16B chunks ----
__device__ __forceinline__ bf16x16 load_a_global(const ushort_t* __restrict__ A,
                                                 int row0, int lda, int k0) {
  int lane = threadIdx.x & 31;
  int half = lane >> 4, r = lane & 15;
  const ushort_t* p = A + (size_t)(row0 + r) * lda + k0 + (half << 3);
  FragU f;
  f.q[0] = *(const u32x4*)(p);        // K = k0 + half*8 + 0..7
  f.q[1] = *(const u32x4*)(p + 16);   // K = k0 + 16 + half*8 + 0..7
  return f.v;
}

// ---- A fragment from LDS bf16 [.. x ld], element-wise, zero-fill K >= kmax ----
__device__ __forceinline__ bf16x16 load_a_lds(const ushort_t* S, int row0, int ld,
                                              int k0, int kmax) {
  int lane = threadIdx.x & 31;
  int half = lane >> 4, r = lane & 15;
  FragU f;
#pragma unroll
  for (int e = 0; e < 16; ++e) {
    int K = k0 + ((e >> 3) << 4) + (half << 3) + (e & 7);
    f.u[e] = (K < kmax) ? S[(row0 + r) * ld + K] : (ushort_t)0;
  }
  return f.v;
}

// ---- B fragment from fragment-linear packed weights: contiguous 32B per lane ----
__device__ __forceinline__ bf16x16 load_b_packed(const ushort_t* __restrict__ Bp,
                                                 int kt, int nt, int ntiles) {
  int lane = threadIdx.x & 31;
  const ushort_t* p = Bp + (((size_t)(kt * ntiles + nt)) << 9) + (lane << 4);
  FragU f;
  f.q[0] = *(const u32x4*)(p);
  f.q[1] = *(const u32x4*)(p + 8);
  return f.v;
}

// ---------------- weight packing: W[N][ldW] (f32) -> fragment-linear bf16 ----------------
__global__ void pack_b_kernel(const float* __restrict__ W, int ldW, int colOff,
                              int Kreal, int Kpad, int Nout, ushort_t* __restrict__ dst) {
  int idx = blockIdx.x * blockDim.x + threadIdx.x;
  int total = Kpad * Nout;
  if (idx >= total) return;
  int frag = idx >> 9;
  int within = idx & 511;
  int lane = within >> 4, e = within & 15;
  int ntiles = Nout >> 4;
  int kt = frag / ntiles, nt = frag % ntiles;
  int c = lane & 15, half = lane >> 4;
  int k = kt * 32 + ((e >> 3) << 4) + half * 8 + (e & 7);      // ISA 16-bit B layout: K = (e/8)*16 + half*8 + e%8
  int n = nt * 16 + c;
  float v = (k < Kreal) ? W[(size_t)n * ldW + colOff + k] : 0.f;
  dst[idx] = f2bf(v);
}

// ---------------- KNN: one block per query point, 64 iterative argmins ----------------
__global__ __launch_bounds__(256) void knn_kernel(const float* __restrict__ P,
                                                  int* __restrict__ idx64) {
  __shared__ float dist[1024];
  __shared__ float rd[256];
  __shared__ int   ri[256];
  int m = blockIdx.x;
  int b = m >> 10, n = m & 1023;
  const float* Px = P + (size_t)b * 3072;
  float px = Px[n], py = Px[1024 + n], pz = Px[2048 + n];
  for (int j = threadIdx.x; j < 1024; j += 256) {
    float dx = Px[j] - px, dy = Px[1024 + j] - py, dz = Px[2048 + j] - pz;
    dist[j] = dx * dx + dy * dy + dz * dz;
  }
  __syncthreads();
  for (int it = 0; it < 64; ++it) {
    float bd = 1e30f; int bi = 0x7FFFFFFF;
    for (int j = threadIdx.x; j < 1024; j += 256) {
      float d = dist[j];
      if (d < bd || (d == bd && j < bi)) { bd = d; bi = j; }
    }
    rd[threadIdx.x] = bd; ri[threadIdx.x] = bi;
    __syncthreads();
    for (int s = 128; s > 0; s >>= 1) {
      if (threadIdx.x < s) {
        float od = rd[threadIdx.x + s]; int oi = ri[threadIdx.x + s];
        if (od < rd[threadIdx.x] || (od == rd[threadIdx.x] && oi < ri[threadIdx.x])) {
          rd[threadIdx.x] = od; ri[threadIdx.x] = oi;
        }
      }
      __syncthreads();
    }
    if (threadIdx.x == 0) {
      idx64[(size_t)m * 64 + it] = ri[0];
      dist[ri[0]] = 1e30f;
    }
    __syncthreads();
  }
}

// ---------------- rotation-invariant local frames ----------------
__global__ void frames_kernel(const float* __restrict__ P, const int* __restrict__ idx64,
                              float* __restrict__ frames) {
  int m = blockIdx.x * blockDim.x + threadIdx.x;
  if (m >= 8192) return;
  int b = m >> 10, n = m & 1023;
  const float* Px = P + (size_t)b * 3072;
  float px = Px[n], py = Px[1024 + n], pz = Px[2048 + n];
  float nr = sqrtf(px * px + py * py + pz * pz) + 1e-8f;
  float a1x = px / nr, a1y = py / nr, a1z = pz / nr;
  float sx = 0.f, sy = 0.f, sz = 0.f;
  for (int j = 0; j < 32; ++j) {
    int id = idx64[(size_t)m * 64 + 2 * j];         // dilated knn
    sx += Px[id] - px; sy += Px[1024 + id] - py; sz += Px[2048 + id] - pz;
  }
  float mx = sx * (1.f / 32.f), my = sy * (1.f / 32.f), mz = sz * (1.f / 32.f);
  float dp = mx * a1x + my * a1y + mz * a1z;
  mx -= dp * a1x; my -= dp * a1y; mz -= dp * a1z;
  float n2 = sqrtf(mx * mx + my * my + mz * mz) + 1e-8f;
  float a2x = mx / n2, a2y = my / n2, a2z = mz / n2;
  float a3x = a1y * a2z - a1z * a2y;
  float a3y = a1z * a2x - a1x * a2z;
  float a3z = a1x * a2y - a1y * a2x;
  float* F = frames + (size_t)m * 9;
  F[0] = a1x; F[1] = a1y; F[2] = a1z;
  F[3] = a2x; F[4] = a2y; F[5] = a2z;
  F[6] = a3x; F[7] = a3y; F[8] = a3z;
}

// ---------------- fused edge MLP: wave-per-point, 3 WMMA stages via LDS ----------------
#define EWAVES 4
__global__ __launch_bounds__(128) void edge_kernel(
    const float* __restrict__ P, const int* __restrict__ idx64,
    const float* __restrict__ frames,
    const ushort_t* __restrict__ W1, const ushort_t* __restrict__ W2,
    const ushort_t* __restrict__ W3,
    const float* __restrict__ b1, const float* __restrict__ b2,
    const float* __restrict__ b3,
    ushort_t* __restrict__ lfeat, int variant) {
  __shared__ ushort_t sA0[EWAVES][32 * 4];
  __shared__ ushort_t sH1[EWAVES][32 * 64];
  __shared__ ushort_t sH2[EWAVES][32 * 128];
  int w = threadIdx.x >> 5, lane = threadIdx.x & 31;
  int m = blockIdx.x * EWAVES + w;                   // point id 0..8191
  int b = m >> 10, n = m & 1023;
  const float* F = frames + (size_t)m * 9;
  float a1x = F[0], a1y = F[1], a1z = F[2];
  float a2x = F[3], a2y = F[4], a2z = F[5];
  float a3x = F[6], a3y = F[7], a3z = F[8];
  const float* Px = P + (size_t)b * 3072;
  float px = Px[n], py = Px[1024 + n], pz = Px[2048 + n];
  int jj = (variant == 0) ? lane : (2 * lane);
  int id = idx64[(size_t)m * 64 + jj];
  float rx = Px[id] - px, ry = Px[1024 + id] - py, rz = Px[2048 + id] - pz;
  sA0[w][lane * 4 + 0] = f2bf(rx * a1x + ry * a1y + rz * a1z);
  sA0[w][lane * 4 + 1] = f2bf(rx * a2x + ry * a2y + rz * a2z);
  sA0[w][lane * 4 + 2] = f2bf(rx * a3x + ry * a3y + rz * a3z);
  sA0[w][lane * 4 + 3] = 0;
  __syncthreads();
  int half = lane >> 4, c = lane & 15;
  // layer 1: 3 -> 64, relu (A fragment reused over 4 output tiles)
  for (int t = 0; t < 2; ++t) {
    bf16x16 af = load_a_lds(&sA0[w][0], t * 16, 4, 0, 3);
    for (int nt = 0; nt < 4; ++nt) {
      f32x8 acc = zero8();
      bf16x16 bfr = load_b_packed(W1, 0, nt, 4);
      acc = wmma_bf16(af, bfr, acc);
      float bi = b1[nt * 16 + c];
#pragma unroll
      for (int i = 0; i < 8; ++i) {
        float v = fmaxf(acc[i] + bi, 0.f);
        sH1[w][(t * 16 + i + half * 8) * 64 + nt * 16 + c] = f2bf(v);
      }
    }
  }
  __syncthreads();
  // layer 2: 64 -> 128, relu (both A k-fragments hoisted, reused over 8 tiles)
  for (int t = 0; t < 2; ++t) {
    bf16x16 a0 = load_a_lds(&sH1[w][0], t * 16, 64, 0, 64);
    bf16x16 a1 = load_a_lds(&sH1[w][0], t * 16, 64, 32, 64);
    for (int nt = 0; nt < 8; ++nt) {
      f32x8 acc = zero8();
      acc = wmma_bf16(a0, load_b_packed(W2, 0, nt, 8), acc);
      acc = wmma_bf16(a1, load_b_packed(W2, 1, nt, 8), acc);
      float bi = b2[nt * 16 + c];
#pragma unroll
      for (int i = 0; i < 8; ++i) {
        float v = fmaxf(acc[i] + bi, 0.f);
        sH2[w][(t * 16 + i + half * 8) * 128 + nt * 16 + c] = f2bf(v);
      }
    }
  }
  __syncthreads();
  // layer 3: 128 -> 128 (no relu) + max over the 32 neighbor rows.
  // A fragments hoisted per row-tile; per-column running max across both tiles.
  float mcol[8];
#pragma unroll
  for (int nt = 0; nt < 8; ++nt) mcol[nt] = -1e30f;
  for (int t = 0; t < 2; ++t) {
    bf16x16 a0 = load_a_lds(&sH2[w][0], t * 16, 128, 0, 128);
    bf16x16 a1 = load_a_lds(&sH2[w][0], t * 16, 128, 32, 128);
    bf16x16 a2 = load_a_lds(&sH2[w][0], t * 16, 128, 64, 128);
    bf16x16 a3 = load_a_lds(&sH2[w][0], t * 16, 128, 96, 128);
    for (int nt = 0; nt < 8; ++nt) {
      f32x8 acc = zero8();
      acc = wmma_bf16(a0, load_b_packed(W3, 0, nt, 8), acc);
      acc = wmma_bf16(a1, load_b_packed(W3, 1, nt, 8), acc);
      acc = wmma_bf16(a2, load_b_packed(W3, 2, nt, 8), acc);
      acc = wmma_bf16(a3, load_b_packed(W3, 3, nt, 8), acc);
      float bi = b3[nt * 16 + c];
#pragma unroll
      for (int i = 0; i < 8; ++i) mcol[nt] = fmaxf(mcol[nt], acc[i] + bi);
    }
  }
#pragma unroll
  for (int nt = 0; nt < 8; ++nt) {
    float v = fmaxf(mcol[nt], __shfl_xor(mcol[nt], 16));   // combine row halves (wave32)
    if (lane < 16)
      lfeat[(size_t)m * 256 + variant * 128 + nt * 16 + lane] = f2bf(v);
  }
}

// ---------- generic WMMA GEMM: wave computes a 16x64 strip (4 accumulators) ----------
// A fragment is loaded once per k-step and fed to 4 WMMAs -> 4x arithmetic intensity.
__global__ __launch_bounds__(256) void gemm_wmma_kernel(
    const ushort_t* __restrict__ A, const ushort_t* __restrict__ Bp,
    const float* __restrict__ bias, float* __restrict__ outF,
    ushort_t* __restrict__ outB, int M, int K, int N, int relu) {
  int wave = blockIdx.x * (blockDim.x >> 5) + (threadIdx.x >> 5);
  int ngrp = N >> 6;                       // groups of 4 n-tiles (64 cols)
  int tiles = (M >> 4) * ngrp;
  if (wave >= tiles) return;
  int tm = wave / ngrp, tg = wave % ngrp;
  int ntile = N >> 4;
  f32x8 acc0 = zero8(), acc1 = zero8(), acc2 = zero8(), acc3 = zero8();
  int ks = K >> 5;
  for (int k = 0; k < ks; ++k) {
    bf16x16 a = load_a_global(A, tm << 4, K, k << 5);
    acc0 = wmma_bf16(a, load_b_packed(Bp, k, tg * 4 + 0, ntile), acc0);
    acc1 = wmma_bf16(a, load_b_packed(Bp, k, tg * 4 + 1, ntile), acc1);
    acc2 = wmma_bf16(a, load_b_packed(Bp, k, tg * 4 + 2, ntile), acc2);
    acc3 = wmma_bf16(a, load_b_packed(Bp, k, tg * 4 + 3, ntile), acc3);
  }
  int lane = threadIdx.x & 31;
  int half = lane >> 4, c = lane & 15;
  f32x8 accs[4] = {acc0, acc1, acc2, acc3};
#pragma unroll
  for (int j = 0; j < 4; ++j) {
    int tn = tg * 4 + j;
    float bi = bias ? bias[(tn << 4) + c] : 0.f;
#pragma unroll
    for (int i = 0; i < 8; ++i) {
      float v = accs[j][i] + bi;
      if (relu) v = fmaxf(v, 0.f);
      size_t o = (size_t)((tm << 4) + i + (half << 3)) * N + (tn << 4) + c;
      if (outF) outF[o] = v;
      if (outB) outB[o] = f2bf(v);
    }
  }
}

// ---------------- decoder helpers ----------------
__global__ void gproj_kernel(const float* __restrict__ dw0, const float* __restrict__ gfeat,
                             float* __restrict__ gproj) {
  int idx = blockIdx.x * blockDim.x + threadIdx.x;
  if (idx >= 8 * 512) return;
  int b = idx >> 9, o = idx & 511;
  const float* w = dw0 + (size_t)o * 2053 + 1029;   // global-feat columns
  const float* g = gfeat + (size_t)b * 1024;
  float s = 0.f;
  for (int k = 0; k < 1024; ++k) s += w[k] * g[k];
  gproj[idx] = s;
}

__global__ void augment_kernel(const float* __restrict__ base, const float* __restrict__ gproj,
                               const float* __restrict__ dw0, const float* __restrict__ db0,
                               const float* __restrict__ grid, const float* __restrict__ P,
                               ushort_t* __restrict__ h1) {
  size_t idx = (size_t)blockIdx.x * blockDim.x + threadIdx.x;
  if (idx >= (size_t)32768 * 512) return;
  int o = (int)(idx & 511);
  size_t row = idx >> 9;
  int b = (int)(row >> 12);
  int f = (int)(row & 4095);
  int n = f >> 2, s = f & 3;
  const float* w = dw0 + (size_t)o * 2053;
  float v = base[((size_t)b * 1024 + n) * 512 + o] + gproj[b * 512 + o] + db0[o];
  v += w[0] * grid[s] + w[1] * grid[4 + s];         // grid columns
  const float* Px = P + (size_t)b * 3072;
  v += w[2] * Px[n] + w[3] * Px[1024 + n] + w[4] * Px[2048 + n];  // point columns
  h1[idx] = f2bf(fmaxf(v, 0.f));
}

__global__ void final_kernel(const ushort_t* __restrict__ h2, const float* __restrict__ W3,
                             const float* __restrict__ b3, const float* __restrict__ P,
                             float* __restrict__ out) {
  int idx = blockIdx.x * blockDim.x + threadIdx.x;
  if (idx >= 8 * 3 * 4096) return;
  int f = idx & 4095;
  int c = (idx >> 12) % 3;
  int b = idx / (3 * 4096);
  const ushort_t* h = h2 + ((size_t)b * 4096 + f) * 512;
  const float* w = W3 + (size_t)c * 512;
  float s = b3[c];
  for (int k = 0; k < 512; ++k) s += bf2f(h[k]) * w[k];
  int n = f >> 2;
  s += P[((size_t)b * 3 + c) * 1024 + n];           // + center
  out[idx] = s;
}

// ---------------- host orchestration ----------------
extern "C" void kernel_launch(void* const* d_in, const int* in_sizes, int n_in,
                              void* d_out, int out_size, void* d_ws, size_t ws_size,
                              hipStream_t stream) {
  (void)in_sizes; (void)n_in; (void)out_size; (void)ws_size;
  const float* gfeat = (const float*)d_in[0];
  const float* P     = (const float*)d_in[1];
  const float* m1w0 = (const float*)d_in[2];
  const float* m1w1 = (const float*)d_in[3];
  const float* m1w2 = (const float*)d_in[4];
  const float* m1b0 = (const float*)d_in[5];
  const float* m1b1 = (const float*)d_in[6];
  const float* m1b2 = (const float*)d_in[7];
  const float* m2w0 = (const float*)d_in[8];
  const float* m2w1 = (const float*)d_in[9];
  const float* m2w2 = (const float*)d_in[10];
  const float* m2b0 = (const float*)d_in[11];
  const float* m2b1 = (const float*)d_in[12];
  const float* m2b2 = (const float*)d_in[13];
  const float* mgw0 = (const float*)d_in[14];
  const float* mgw1 = (const float*)d_in[15];
  const float* mgw2 = (const float*)d_in[16];
  const float* mgb0 = (const float*)d_in[17];
  const float* mgb1 = (const float*)d_in[18];
  const float* mgb2 = (const float*)d_in[19];
  const float* dw0  = (const float*)d_in[20];
  const float* dw1  = (const float*)d_in[21];
  const float* dw2  = (const float*)d_in[22];
  const float* db0  = (const float*)d_in[23];
  const float* db1  = (const float*)d_in[24];
  const float* db2  = (const float*)d_in[25];
  const float* grid = (const float*)d_in[26];
  float* out = (float*)d_out;

  char* ws = (char*)d_ws;
  size_t off = 0;
  auto alloc = [&](size_t bytes) -> char* {
    char* p = ws + off;
    off = (off + bytes + 255) & ~(size_t)255;
    return p;
  };
  int*      idx64 = (int*)alloc((size_t)8192 * 64 * 4);
  float*    frames = (float*)alloc((size_t)8192 * 9 * 4);
  ushort_t* lfeat = (ushort_t*)alloc((size_t)8192 * 256 * 2);
  ushort_t* mH1   = (ushort_t*)alloc((size_t)8192 * 512 * 2);
  ushort_t* mH2   = (ushort_t*)alloc((size_t)8192 * 1024 * 2);
  ushort_t* locf  = (ushort_t*)alloc((size_t)8192 * 1024 * 2);
  float*    base  = (float*)alloc((size_t)8192 * 512 * 4);
  float*    gprojB = (float*)alloc((size_t)8 * 512 * 4);
  ushort_t* h1 = (ushort_t*)alloc((size_t)32768 * 512 * 2);
  ushort_t* h2 = (ushort_t*)alloc((size_t)32768 * 512 * 2);
  ushort_t* pe1a = (ushort_t*)alloc(32 * 64 * 2);
  ushort_t* pe1b = (ushort_t*)alloc(64 * 128 * 2);
  ushort_t* pe1c = (ushort_t*)alloc(128 * 128 * 2);
  ushort_t* pe2a = (ushort_t*)alloc(32 * 64 * 2);
  ushort_t* pe2b = (ushort_t*)alloc(64 * 128 * 2);
  ushort_t* pe2c = (ushort_t*)alloc(128 * 128 * 2);
  ushort_t* pm1  = (ushort_t*)alloc((size_t)256 * 512 * 2);
  ushort_t* pm2  = (ushort_t*)alloc((size_t)512 * 1024 * 2);
  ushort_t* pm3  = (ushort_t*)alloc((size_t)1024 * 1024 * 2);
  ushort_t* pWl  = (ushort_t*)alloc((size_t)1024 * 512 * 2);
  ushort_t* pW2d = (ushort_t*)alloc((size_t)512 * 512 * 2);

  auto pack = [&](const float* W, int ldW, int colOff, int Kreal, int Kpad, int Nout,
                  ushort_t* dst) {
    int total = Kpad * Nout;
    pack_b_kernel<<<(total + 255) / 256, 256, 0, stream>>>(W, ldW, colOff, Kreal, Kpad,
                                                           Nout, dst);
  };
  pack(m1w0, 3, 0, 3, 32, 64, pe1a);
  pack(m1w1, 64, 0, 64, 64, 128, pe1b);
  pack(m1w2, 128, 0, 128, 128, 128, pe1c);
  pack(m2w0, 3, 0, 3, 32, 64, pe2a);
  pack(m2w1, 64, 0, 64, 64, 128, pe2b);
  pack(m2w2, 128, 0, 128, 128, 128, pe2c);
  pack(mgw0, 256, 0, 256, 256, 512, pm1);
  pack(mgw1, 512, 0, 512, 512, 1024, pm2);
  pack(mgw2, 1024, 0, 1024, 1024, 1024, pm3);
  pack(dw0, 2053, 5, 1024, 1024, 512, pWl);   // local-feat columns of dec layer 1
  pack(dw1, 512, 0, 512, 512, 512, pW2d);

  knn_kernel<<<8192, 256, 0, stream>>>(P, idx64);
  frames_kernel<<<32, 256, 0, stream>>>(P, idx64, frames);
  edge_kernel<<<2048, 128, 0, stream>>>(P, idx64, frames, pe1a, pe1b, pe1c,
                                        m1b0, m1b1, m1b2, lfeat, 0);
  edge_kernel<<<2048, 128, 0, stream>>>(P, idx64, frames, pe2a, pe2b, pe2c,
                                        m2b0, m2b1, m2b2, lfeat, 1);

  auto gemm = [&](const ushort_t* A, const ushort_t* Bp, const float* bias, float* oF,
                  ushort_t* oB, int M, int K, int N, int relu) {
    int tiles = (M / 16) * (N / 64);
    gemm_wmma_kernel<<<(tiles + 7) / 8, 256, 0, stream>>>(A, Bp, bias, oF, oB, M, K, N,
                                                          relu);
  };
  gemm(lfeat, pm1, mgb0, nullptr, mH1, 8192, 256, 512, 1);
  gemm(mH1, pm2, mgb1, nullptr, mH2, 8192, 512, 1024, 1);
  gemm(mH2, pm3, mgb2, nullptr, locf, 8192, 1024, 1024, 0);
  gemm(locf, pWl, nullptr, base, nullptr, 8192, 1024, 512, 0);

  gproj_kernel<<<16, 256, 0, stream>>>(dw0, gfeat, gprojB);
  augment_kernel<<<65536, 256, 0, stream>>>(base, gprojB, dw0, db0, grid, P, h1);
  gemm(h1, pW2d, db1, nullptr, h2, 32768, 512, 512, 1);
  final_kernel<<<384, 256, 0, stream>>>(h2, dw2, db2, P, out);
}
